// Mamba_Layer_18923625906802
// MI455X (gfx1250) — matrix operand order repaired and checked by reading, not compiled, not run on hardware
//
#include <hip/hip_runtime.h>
#include <hip/hip_bf16.h>

#define B_ 2
#define L_ 4096
#define DM 256
#define DI 512
#define DS 16
#define DR 16
#define NROW (B_ * L_)   // 8192

typedef __attribute__((ext_vector_type(16))) __bf16  v16bf;
typedef __attribute__((ext_vector_type(8)))  float   v8f;
typedef __attribute__((ext_vector_type(4)))  unsigned int uint4v;

// ---------- helpers ----------
__device__ __forceinline__ unsigned short f2bf(float f) {
    unsigned int u = __float_as_uint(f);
    u += 0x7FFFu + ((u >> 16) & 1u);        // round-to-nearest-even
    return (unsigned short)(u >> 16);
}
__device__ __forceinline__ float silu(float v) {
    return v / (1.0f + __expf(-v));
}

// ---------- prep kernels ----------
__global__ void cvt_bf16_kernel(const float* __restrict__ src,
                                unsigned short* __restrict__ dst, int n) {
    int i = blockIdx.x * blockDim.x + threadIdx.x;
    if (i < n) dst[i] = f2bf(src[i]);
}

// src is [Kdim, Ndim] row-major; dst is [Ndim, Kdim] (each output row = one column)
__global__ void transpose_cvt_kernel(const float* __restrict__ src,
                                     unsigned short* __restrict__ dst,
                                     int Kdim, int Ndim) {
    int i = blockIdx.x * blockDim.x + threadIdx.x;
    if (i >= Kdim * Ndim) return;
    int nn = i / Kdim;
    int kk = i - nn * Kdim;
    dst[i] = f2bf(src[(size_t)kk * Ndim + nn]);
}

__global__ void aneg_kernel(const float* __restrict__ A_log,
                            float* __restrict__ Aneg, int n) {
    int i = blockIdx.x * blockDim.x + threadIdx.x;
    if (i < n) Aneg[i] = -__expf(A_log[i]);
}

// ---------- WMMA GEMM: C[M,N] = A[M,K](bf16,row-major) * Bt[N,K](bf16)^T ----------
// One wave computes a 16 x (NT*16) strip; the A fragment is loaded once per
// K-step and reused across NT wmma issues. K is a compile-time constant so all
// B-tile / k-step displacements fold into the 24-bit immediate offset of
// global_load_b128 (only two 64-bit base pointers live in the loop -> no
// address spills). The K-loop is unrolled 2x with named fragment stages so
// loads of one stage overlap the wmmas of the other (partial s_wait_loadcnt).
// M mult of 16, N mult of NT*16, K mult of 64.
template <int NT, int K>
__global__ void wmma_gemm_bf16(const unsigned short* __restrict__ A,
                               const unsigned short* __restrict__ Bt,
                               float* __restrict__ C,
                               int M, int N) {
    int lane = threadIdx.x & 31;
    int wave = threadIdx.x >> 5;
    int gid  = blockIdx.x * (blockDim.x >> 5) + wave;   // strip index
    int tilesM  = M >> 4;
    int tilesNG = N / (16 * NT);
    if (gid >= tilesM * tilesNG) return;   // wave-uniform: EXEC stays all-1 for WMMA
    int tm = (gid / tilesNG) << 4;
    int tn = (gid - (gid / tilesNG) * tilesNG) * (16 * NT);

    int half = lane >> 4;                  // 0: lanes 0-15, 1: lanes 16-31
    int mn   = lane & 15;

    const unsigned short* Arow  = A  + (size_t)(tm + mn) * K;   // lane = row M
    const unsigned short* Brow0 = Bt + (size_t)(tn + mn) * K;   // lane = col N (tile 0)

    union frag { v16bf v; uint4v q[2]; };
    frag a0, a1;
    frag b0[NT], b1[NT];

    // A 16x32 layout (ISA 7.12.2): lanes 0-15 hold K=[k..k+7],[k+16..k+23];
    //                              lanes 16-31 hold K=[k+8..k+15],[k+24..k+31]
    // B 32x16: lanes 0-15 hold K=[k..k+7],[k+16..k+23] of their column; 16-31 K=[k+8..k+15],[k+24..k+31]
    auto load_stage = [&](frag& fa, frag (&fb)[NT], int k) {
        fa.q[0] = *(const uint4v*)(Arow + k + half * 8);
        fa.q[1] = *(const uint4v*)(Arow + k + 16 + half * 8);
#pragma unroll
        for (int t = 0; t < NT; ++t) {
            // t*16*K is a compile-time constant -> immediate offset, single base ptr
            fb[t].q[0] = *(const uint4v*)(Brow0 + t * 16 * K + k + half * 8);
            fb[t].q[1] = *(const uint4v*)(Brow0 + t * 16 * K + k + 16 + half * 8);
        }
    };

    v8f c[NT];
#pragma unroll
    for (int t = 0; t < NT; ++t) c[t] = {};

    load_stage(a0, b0, 0);                  // prologue
#pragma unroll
    for (int k = 0; k < K; k += 64) {       // K % 64 == 0, compile-time trip count
        load_stage(a1, b1, k + 32);         // overlaps wmmas on stage 0
#pragma unroll
        for (int t = 0; t < NT; ++t)
            c[t] = __builtin_amdgcn_wmma_f32_16x16x32_bf16(false, a0.v, false, b0[t].v,
                                                           (short)0, c[t], false, false);
        if (k + 64 < K)
            load_stage(a0, b0, k + 64);     // overlaps wmmas on stage 1
#pragma unroll
        for (int t = 0; t < NT; ++t)
            c[t] = __builtin_amdgcn_wmma_f32_16x16x32_bf16(false, a1.v, false, b1[t].v,
                                                           (short)0, c[t], false, false);
    }

    // C/D layout: VGPR r -> row tm + r + half*8, col tn + t*16 + mn
#pragma unroll
    for (int t = 0; t < NT; ++t) {
        float* Crow = C + (size_t)(tm + half * 8) * N + tn + t * 16 + mn;
#pragma unroll
        for (int r = 0; r < 8; ++r) Crow[(size_t)r * N] = c[t][r];
    }
}

// ---------- depthwise causal conv (4 taps) + SiLU ----------
__global__ void conv_silu_kernel(const float* __restrict__ xz,
                                 const float* __restrict__ conv_w,
                                 const float* __restrict__ conv_b,
                                 float* __restrict__ uconv,
                                 unsigned short* __restrict__ uconv_bf) {
    int idx = blockIdx.x * blockDim.x + threadIdx.x;   // over B*L*DI
    if (idx >= NROW * DI) return;
    int d  = idx & (DI - 1);
    int bl = idx >> 9;                                  // DI = 512
    int l  = bl & (L_ - 1);
    float acc = conv_b[d];
#pragma unroll
    for (int j = 0; j < 4; ++j) {
        int lt = l - 3 + j;
        if (lt >= 0)
            acc += conv_w[d * 4 + j] * xz[(size_t)(bl - (3 - j)) * (2 * DI) + d];
    }
    float s = silu(acc);
    uconv[idx]    = s;
    uconv_bf[idx] = f2bf(s);
}

// ---------- dt_proj + softplus ----------
__global__ void delta_kernel(const float* __restrict__ dbl,
                             const float* __restrict__ dt_w,
                             const float* __restrict__ dt_b,
                             float* __restrict__ delta) {
    int idx = blockIdx.x * blockDim.x + threadIdx.x;   // over NROW*DI
    if (idx >= NROW * DI) return;
    int d   = idx & (DI - 1);
    int row = idx >> 9;
    float acc = dt_b[d];
#pragma unroll
    for (int r = 0; r < DR; ++r)
        acc += dbl[(size_t)row * 48 + r] * dt_w[(size_t)r * DI + d];
    delta[idx] = (acc > 20.0f) ? acc : log1pf(__expf(acc));
}

// ---------- selective scan (serial over L; one wave = 2 channels x 16 states) ----------
__global__ void scan_kernel(const float* __restrict__ delta,
                            const float* __restrict__ uconv,
                            const float* __restrict__ dbl,
                            const float* __restrict__ Aneg,
                            const float* __restrict__ Dvec,
                            float* __restrict__ ys) {
    int lane = threadIdx.x & 31;
    int wave = (blockIdx.x * blockDim.x + threadIdx.x) >> 5;   // 0 .. B*DI/2-1
    if (wave >= B_ * DI / 2) return;
    int b    = wave / (DI / 2);
    int pair = wave - b * (DI / 2);
    int half = lane >> 4;
    int n    = lane & 15;
    int d    = pair * 2 + half;

    float Ac = Aneg[d * DS + n];
    float Dd = Dvec[d];
    float h  = 0.0f;
    size_t base = (size_t)b * L_;
    for (int l = 0; l < L_; ++l) {
        size_t rl = base + l;
        float dl = delta[rl * DI + d];
        float ud = uconv[rl * DI + d];
        const float* dr = dbl + rl * 48;
        float Bn = dr[16 + n];
        float Cn = dr[32 + n];
        h = __expf(dl * Ac) * h + dl * Bn * ud;
        float p = h * Cn;
        p += __shfl_xor(p, 1);
        p += __shfl_xor(p, 2);
        p += __shfl_xor(p, 4);
        p += __shfl_xor(p, 8);       // xor of bits 0-3 stays within each 16-lane half
        if (n == 0) ys[rl * DI + d] = p + ud * Dd;
    }
}

// ---------- gate: y * silu(z) -> bf16 ----------
__global__ void gate_kernel(const float* __restrict__ ys,
                            const float* __restrict__ xz,
                            unsigned short* __restrict__ ybf) {
    int idx = blockIdx.x * blockDim.x + threadIdx.x;   // over NROW*DI
    if (idx >= NROW * DI) return;
    int d  = idx & (DI - 1);
    int bl = idx >> 9;
    float z = xz[(size_t)bl * (2 * DI) + DI + d];
    ybf[idx] = f2bf(ys[idx] * silu(z));
}

// ---------- residual + layernorm (one wave per row of 256) ----------
__global__ void ln_kernel(const float* __restrict__ x,
                          const float* __restrict__ mm,
                          const float* __restrict__ gamma,
                          const float* __restrict__ beta,
                          float* __restrict__ out) {
    int lane = threadIdx.x & 31;
    int row  = (blockIdx.x * blockDim.x + threadIdx.x) >> 5;
    if (row >= NROW) return;
    const float* xr = x  + (size_t)row * DM;
    const float* mr = mm + (size_t)row * DM;
    float v[8], s = 0.0f, s2 = 0.0f;
#pragma unroll
    for (int i = 0; i < 8; ++i) {
        int c = lane * 8 + i;
        float t = xr[c] + mr[c];
        v[i] = t; s += t; s2 += t * t;
    }
    for (int off = 1; off < 32; off <<= 1) {
        s  += __shfl_xor(s,  off);
        s2 += __shfl_xor(s2, off);
    }
    float mean = s * (1.0f / DM);
    float var  = s2 * (1.0f / DM) - mean * mean;
    float inv  = rsqrtf(var + 1e-5f);
    float* orow = out + (size_t)row * DM;
#pragma unroll
    for (int i = 0; i < 8; ++i) {
        int c = lane * 8 + i;
        orow[c] = (v[i] - mean) * inv * gamma[c] + beta[c];
    }
}

// ---------- host ----------
extern "C" void kernel_launch(void* const* d_in, const int* in_sizes, int n_in,
                              void* d_out, int out_size, void* d_ws, size_t ws_size,
                              hipStream_t stream) {
    const float* x        = (const float*)d_in[0];   // [B,L,256]
    const float* in_w     = (const float*)d_in[1];   // [256,1024]
    const float* conv_w   = (const float*)d_in[2];   // [512,4]
    const float* conv_b   = (const float*)d_in[3];   // [512]
    const float* xproj_w  = (const float*)d_in[4];   // [512,48]
    const float* dt_w     = (const float*)d_in[5];   // [16,512]
    const float* dt_b     = (const float*)d_in[6];   // [512]
    const float* A_log    = (const float*)d_in[7];   // [512,16]
    const float* Dvec     = (const float*)d_in[8];   // [512]
    const float* out_w    = (const float*)d_in[9];   // [512,256]
    const float* ln_g     = (const float*)d_in[10];  // [256]
    const float* ln_b     = (const float*)d_in[11];  // [256]
    float* out = (float*)d_out;

    char* ws = (char*)d_ws;
    size_t off = 0;
    auto alloc = [&](size_t bytes) -> char* {
        char* p = ws + off;
        off = (off + bytes + 255) & ~(size_t)255;
        return p;
    };
    unsigned short* Xb    = (unsigned short*)alloc((size_t)NROW * DM * 2);       // x bf16
    unsigned short* Wib   = (unsigned short*)alloc((size_t)(2 * DI) * DM * 2);   // in_proj^T  [1024,256]
    unsigned short* Wxb   = (unsigned short*)alloc((size_t)48 * DI * 2);         // x_proj^T   [48,512]
    unsigned short* Wob   = (unsigned short*)alloc((size_t)DM * DI * 2);         // out_proj^T [256,512]
    float*          Aneg  = (float*)alloc((size_t)DI * DS * 4);
    float*          xz    = (float*)alloc((size_t)NROW * 2 * DI * 4);            // [8192,1024]
    float*          uconv = (float*)alloc((size_t)NROW * DI * 4);
    unsigned short* ucb   = (unsigned short*)alloc((size_t)NROW * DI * 2);
    float*          dbl   = (float*)alloc((size_t)NROW * 48 * 4);
    float*          delta = (float*)alloc((size_t)NROW * DI * 4);
    float*          ysb   = (float*)alloc((size_t)NROW * DI * 4);
    unsigned short* ybf   = (unsigned short*)alloc((size_t)NROW * DI * 2);
    float*          mm    = (float*)alloc((size_t)NROW * DM * 4);
    (void)ws_size; (void)n_in; (void)in_sizes; (void)out_size;

    const int T = 256;
    auto blocks = [](int n, int t) { return (n + t - 1) / t; };

    // 1. prep / precision conversion
    cvt_bf16_kernel<<<blocks(NROW * DM, T), T, 0, stream>>>(x, Xb, NROW * DM);
    transpose_cvt_kernel<<<blocks(DM * 2 * DI, T), T, 0, stream>>>(in_w,    Wib, DM, 2 * DI);
    transpose_cvt_kernel<<<blocks(DI * 48,     T), T, 0, stream>>>(xproj_w, Wxb, DI, 48);
    transpose_cvt_kernel<<<blocks(DI * DM,     T), T, 0, stream>>>(out_w,   Wob, DI, DM);
    aneg_kernel<<<blocks(DI * DS, T), T, 0, stream>>>(A_log, Aneg, DI * DS);

    // 2. in_proj: [8192,256] x [256,1024] -> xz   (strips of 16x64, NT=4, K=256)
    {
        int strips = (NROW / 16) * ((2 * DI) / 64);       // 8192 waves
        wmma_gemm_bf16<4, DM><<<strips / 8, 256, 0, stream>>>(Xb, Wib, xz, NROW, 2 * DI);
    }
    // 3. conv + silu
    conv_silu_kernel<<<blocks(NROW * DI, T), T, 0, stream>>>(xz, conv_w, conv_b, uconv, ucb);
    // 4. x_proj: [8192,512] x [512,48] -> dbl     (strips of 16x48, NT=3, K=512)
    {
        int strips = (NROW / 16) * (48 / 48);             // 512 waves
        wmma_gemm_bf16<3, DI><<<strips / 8, 256, 0, stream>>>(ucb, Wxb, dbl, NROW, 48);
    }
    // 5. dt_proj + softplus
    delta_kernel<<<blocks(NROW * DI, T), T, 0, stream>>>(dbl, dt_w, dt_b, delta);
    // 6. selective scan (+ u*D folded in)
    {
        int waves = B_ * DI / 2;                          // 512
        scan_kernel<<<waves / 8, 256, 0, stream>>>(delta, uconv, dbl, Aneg, Dvec, ysb);
    }
    // 7. gating
    gate_kernel<<<blocks(NROW * DI, T), T, 0, stream>>>(ysb, xz, ybf);
    // 8. out_proj: [8192,512] x [512,256] -> mm   (strips of 16x64, NT=4, K=512)
    {
        int strips = (NROW / 16) * (DM / 64);             // 2048 waves
        wmma_gemm_bf16<4, DI><<<strips / 8, 256, 0, stream>>>(ybf, Wob, mm, NROW, DM);
    }
    // 9. residual + layernorm
    ln_kernel<<<(NROW / 8), 256, 0, stream>>>(x, mm, ln_g, ln_b, out);
}
